// CrossAttention_70987219468614
// MI455X (gfx1250) — matrix operand order repaired and checked by reading, not compiled, not run on hardware
//
#include <hip/hip_runtime.h>
#include <hip/hip_bf16.h>

// Problem constants (fixed by the reference)
#define B_SZ   4
#define NQ_    4096
#define NK_    1024
#define QDIM_  512
#define CDIM_  768
#define HEADS_ 8
#define DHEAD_ 64
#define INNER_ 512

typedef __attribute__((ext_vector_type(16))) _Float16 v16h;
typedef __attribute__((ext_vector_type(8)))  _Float16 v8h;
typedef __attribute__((ext_vector_type(4)))  _Float16 v4h;
typedef __attribute__((ext_vector_type(8)))  float    v8f;

#define CAT16(lo, hh) __builtin_shufflevector((lo), (hh), 0, 1, 2, 3, 4, 5, 6, 7, \
                                              8, 9, 10, 11, 12, 13, 14, 15)

// ---------------------------------------------------------------------------
// WMMA wrapper: D = A(16x32 f16) * B(32x16 f16) + C(16x16 f32)
// ---------------------------------------------------------------------------
__device__ __forceinline__ v8f wmma16(v16h a, v16h b, v8f c) {
  return __builtin_amdgcn_wmma_f32_16x16x32_f16(
      /*neg_a=*/false, a, /*neg_b=*/false, b,
      /*c_mod=*/(short)0, c, /*reuse_a=*/false, /*reuse_b=*/false);
}

// ---------------------------------------------------------------------------
// Fragment loaders (CDNA5 ISA 7.12.2 layouts, wave32)
// A frag 16x32 f16: lane(m = lane&15, hi = lane>>4) holds K = hi*8+0..7 and
// 16+hi*8+0..7 -> two contiguous 16B chunks.
// B frag 32x16 f16: lane(n = lane&15, hi) holds K = hi*8+0..7 and 16+hi*8+0..7.
// C/D 16x16 f32: VGPR r -> row r + hi*8, col = lane&15.
// ---------------------------------------------------------------------------
__device__ __forceinline__ v16h a_frag_f16g(const _Float16* __restrict__ base,
                                            int ld, int m0, int k0) {
  const int lane = threadIdx.x & 31;
  const int m = m0 + (lane & 15), hi = lane >> 4;
  const _Float16* p = base + (size_t)m * ld + k0;
  v8h lo = *(const v8h*)(p + hi * 8);
  v8h hh = *(const v8h*)(p + 16 + hi * 8);
  return CAT16(lo, hh);
}

__device__ __forceinline__ v16h a_frag_f32g(const float* __restrict__ base,
                                            int ld, int m0, int k0) {
  const int lane = threadIdx.x & 31;
  const int m = m0 + (lane & 15), hi = lane >> 4;
  const float* p = base + (size_t)m * ld + k0;
  v8f lo = *(const v8f*)(p + hi * 8);
  v8f hh = *(const v8f*)(p + 16 + hi * 8);
  v8h l2 = __builtin_convertvector(lo, v8h);
  v8h h2 = __builtin_convertvector(hh, v8h);
  return CAT16(l2, h2);
}

__device__ __forceinline__ v16h a_frag_lds(const _Float16* p, int stride) {
  const int lane = threadIdx.x & 31;
  const int m = lane & 15, hi = lane >> 4;
  const _Float16* row = p + m * stride;
  v8h lo = *(const v8h*)(row + hi * 8);
  v8h hh = *(const v8h*)(row + 16 + hi * 8);
  return CAT16(lo, hh);
}

// B frag where B's column n is contiguous in memory along K (e.g. K-matrix
// rows for S = Q*K^T): two 16B chunks per lane, direct from global.
__device__ __forceinline__ v16h b_frag_colmajor_g(const _Float16* __restrict__ base,
                                                  int ld, int n0, int k0) {
  const int lane = threadIdx.x & 31;
  const _Float16* p = base + (size_t)(n0 + (lane & 15)) * ld + k0 + (lane >> 4) * 8;
  v8h lo = *(const v8h*)p;
  v8h hh = *(const v8h*)(p + 16);
  return CAT16(lo, hh);
}

// B frag from LDS stored [n][K] with padded stride (halves)
__device__ __forceinline__ v16h b_frag_lds(const _Float16* lds, int n0, int stride) {
  const int lane = threadIdx.x & 31;
  const _Float16* p = lds + (n0 + (lane & 15)) * stride + (lane >> 4) * 8;
  v8h lo = *(const v8h*)p;
  v8h hh = *(const v8h*)(p + 16);
  return CAT16(lo, hh);
}

// ---------------------------------------------------------------------------
// GEMM: C[M,N] = A[M,K] * W[K,N] (+bias). A f32 or f16; W f32 -> f16 LDS panel.
// Block: 256 threads = 8 waves (4 along M x 2 along N); tile 128x64.
// Each wave: 32x32 = 2x2 WMMA tiles, K stepped by 32.
// W panel is software-pipelined: next panel's global loads are issued right
// after the barrier so they overlap this iteration's WMMA work.
// ---------------------------------------------------------------------------
#define LDSS 40  // padded LDS row stride in halves (80B) to spread banks

template <bool A_F16, bool HAS_BIAS, bool OUT_F32>
__global__ __launch_bounds__(256) void gemm_wmma_kernel(
    const void* __restrict__ Aptr, const float* __restrict__ W,
    const float* __restrict__ bias, void* __restrict__ Cptr,
    int M, int K, int N) {
  __shared__ _Float16 Bs[64 * LDSS];  // W panel, [n][k] f16

  const int tid  = threadIdx.x;
  const int wave = tid >> 5, lane = tid & 31;
  const int wm = wave & 3, wn = wave >> 2;
  const int m_blk = blockIdx.x * 128;
  const int n_blk = blockIdx.y * 64;

  const int n  = tid & 31;          // coalesced across a wave
  const int k4 = (tid >> 5) * 4;    // 0,4,...,28

  // ---- preload first W panel into registers ----
  float w[2][4];
#pragma unroll
  for (int nn = 0; nn < 2; ++nn)
#pragma unroll
    for (int i = 0; i < 4; ++i)
      w[nn][i] = W[(size_t)(k4 + i) * N + n_blk + nn * 32 + n];

  v8f acc[2][2] = {};

  for (int k0 = 0; k0 < K; k0 += 32) {
    // ---- commit current panel to LDS: Bs[n][k], packed b64 stores ----
#pragma unroll
    for (int nn = 0; nn < 2; ++nn) {
      v4h pk;
#pragma unroll
      for (int i = 0; i < 4; ++i) pk[i] = (_Float16)w[nn][i];
      *(v4h*)&Bs[(nn * 32 + n) * LDSS + k4] = pk;
    }
    __syncthreads();

    // ---- prefetch next panel into registers (overlaps WMMAs below) ----
    if (k0 + 32 < K) {
#pragma unroll
      for (int nn = 0; nn < 2; ++nn)
#pragma unroll
        for (int i = 0; i < 4; ++i)
          w[nn][i] = W[(size_t)(k0 + 32 + k4 + i) * N + n_blk + nn * 32 + n];
    }

    // ---- A fragments straight from global ----
    v16h a0, a1;
    if (A_F16) {
      const _Float16* A = (const _Float16*)Aptr;
      a0 = a_frag_f16g(A, K, m_blk + wm * 32, k0);
      a1 = a_frag_f16g(A, K, m_blk + wm * 32 + 16, k0);
    } else {
      const float* A = (const float*)Aptr;
      a0 = a_frag_f32g(A, K, m_blk + wm * 32, k0);
      a1 = a_frag_f32g(A, K, m_blk + wm * 32 + 16, k0);
    }

    v16h b0 = b_frag_lds(Bs, wn * 32, LDSS);
    v16h b1 = b_frag_lds(Bs, wn * 32 + 16, LDSS);

    acc[0][0] = wmma16(a0, b0, acc[0][0]);
    acc[0][1] = wmma16(a0, b1, acc[0][1]);
    acc[1][0] = wmma16(a1, b0, acc[1][0]);
    acc[1][1] = wmma16(a1, b1, acc[1][1]);
    __syncthreads();  // protect Bs before next commit
  }

  // ---- write out (C layout: VGPR r -> row r + hi*8, col lane&15) ----
  const int hi = lane >> 4, nl = lane & 15;
#pragma unroll
  for (int i = 0; i < 2; ++i) {
#pragma unroll
    for (int j = 0; j < 2; ++j) {
      const int col  = n_blk + wn * 32 + j * 16 + nl;
      const int row0 = m_blk + wm * 32 + i * 16 + hi * 8;
      const float bv = HAS_BIAS ? bias[col] : 0.0f;
#pragma unroll
      for (int r = 0; r < 8; ++r) {
        const float val = acc[i][j][r] + bv;
        if (OUT_F32)
          ((float*)Cptr)[(size_t)(row0 + r) * N + col] = val;
        else
          ((_Float16*)Cptr)[(size_t)(row0 + r) * N + col] = (_Float16)val;
      }
    }
  }
}

// ---------------------------------------------------------------------------
// Fused flash attention: O = softmax(Q K^T / sqrt(D)) V, per (b, h).
// Block: 128 threads = 4 waves; each wave owns 16 query rows (64/block).
// Key tile = 64 positions/iter: 8 S-WMMAs + 8 PV-WMMAs per iteration.
// The softmax scale (1/sqrt(D) * log2 e) is pre-folded into the Q fragments,
// so WMMA scores land directly in exp2-domain units and the softmax consumes
// accumulator registers in place (no repack/scale VALU after the WMMA chain).
// ---------------------------------------------------------------------------
#define KT   64
#define VSTR 72  // padded stride (halves) for Vt / Ps rows

__global__ __launch_bounds__(128) void attn_kernel(
    const _Float16* __restrict__ Q,   // [B*NQ, INNER]
    const _Float16* __restrict__ Kc,  // [B*NK, INNER]
    const _Float16* __restrict__ Vc,  // [B*NK, INNER]
    _Float16* __restrict__ O) {       // [B*NQ, INNER]
  __shared__ _Float16 Vt[64 * VSTR];      // V tile transposed: [d][kk]
  __shared__ _Float16 Ps[4][16 * VSTR];   // per-wave P tile: [m][kk]

  const int tid  = threadIdx.x;
  const int wave = tid >> 5, lane = tid & 31;
  const int hi = lane >> 4, nl = lane & 15;
  const int bh = blockIdx.y;
  const int b  = bh >> 3, h = bh & 7;
  const int q0 = blockIdx.x * 64;

  const _Float16* Qb = Q  + (size_t)b * NQ_ * INNER_ + h * DHEAD_;
  const _Float16* Kb = Kc + (size_t)b * NK_ * INNER_ + h * DHEAD_;
  const _Float16* Vb = Vc + (size_t)b * NQ_ * 0 + (size_t)b * NK_ * INNER_ + h * DHEAD_;

  // Q fragments for this wave's 16 rows (both 32-wide d-chunks), kept in regs.
  // Pre-scale by 1/sqrt(64) * log2(e) so S-WMMA outputs are exp2-domain ready.
  const _Float16 qsc = (_Float16)(0.125f * 1.44269504088896f);
  v16h qa0 = a_frag_f16g(Qb, INNER_, q0 + wave * 16, 0)  * qsc;
  v16h qa1 = a_frag_f16g(Qb, INNER_, q0 + wave * 16, 32) * qsc;

  float m_run[8], l_run[8];
#pragma unroll
  for (int r = 0; r < 8; ++r) { m_run[r] = -3.0e38f; l_run[r] = 0.0f; }
  v8f oacc[4] = {};  // 16 x 64 output accumulator (4 d-tiles)

  // ---- preload first V tile into registers (software pipeline) ----
  const int kk4 = (tid & 15) * 4;   // 0..60
  const int d8  = (tid >> 4) * 8;   // 0..56
  v8h vr[4];
#pragma unroll
  for (int i = 0; i < 4; ++i)
    vr[i] = *(const v8h*)(Vb + (size_t)(kk4 + i) * INNER_ + d8);

  for (int kt = 0; kt < NK_; kt += KT) {
    // ---- commit V tile transposed: Vt[d][kk] = V[kt+kk][h*64+d] ----
    __syncthreads();  // readers of the previous tile are done
#pragma unroll
    for (int j = 0; j < 8; ++j) {
      v4h pk;
#pragma unroll
      for (int i = 0; i < 4; ++i) pk[i] = vr[i][j];
      *(v4h*)&Vt[(d8 + j) * VSTR + kk4] = pk;  // ds_store_b64
    }
    __syncthreads();

    // ---- prefetch next V tile into registers (overlaps all compute) ----
    if (kt + KT < NK_) {
#pragma unroll
      for (int i = 0; i < 4; ++i)
        vr[i] = *(const v8h*)(Vb + (size_t)(kt + KT + kk4 + i) * INNER_ + d8);
      __builtin_prefetch(Kb + (size_t)(kt + KT + (tid & 31)) * INNER_, 0, 0);
    }

    // ---- S = Q K^T : 16 x 64 scores; all K frags first, then WMMA chain ----
    v16h kf[4][2];
#pragma unroll
    for (int t = 0; t < 4; ++t) {
      kf[t][0] = b_frag_colmajor_g(Kb, INNER_, kt + t * 16, 0);
      kf[t][1] = b_frag_colmajor_g(Kb, INNER_, kt + t * 16, 32);
    }
    v8f s[4];
#pragma unroll
    for (int t = 0; t < 4; ++t) {
      v8f z = {};
      s[t] = wmma16(qa0, kf[t][0], z);
      s[t] = wmma16(qa1, kf[t][1], s[t]);
    }

    // ---- online softmax (exp2 domain); row r lives at m = hi*8 + r.
    // Scores are already scaled: consume accumulators in place. ----
#pragma unroll
    for (int r = 0; r < 8; ++r) {
      float mx = fmaxf(fmaxf(s[0][r], s[1][r]), fmaxf(s[2][r], s[3][r]));
#pragma unroll
      for (int off = 1; off < 16; off <<= 1)
        mx = fmaxf(mx, __shfl_xor(mx, off, 32));
      const float m_new = fmaxf(m_run[r], mx);
      const float alpha = exp2f(m_run[r] - m_new);
      float e[4], rs = 0.0f;
#pragma unroll
      for (int t = 0; t < 4; ++t) { e[t] = exp2f(s[t][r] - m_new); rs += e[t]; }
#pragma unroll
      for (int off = 1; off < 16; off <<= 1)
        rs += __shfl_xor(rs, off, 32);
      l_run[r] = l_run[r] * alpha + rs;
      m_run[r] = m_new;
#pragma unroll
      for (int t = 0; t < 4; ++t) oacc[t][r] *= alpha;
      // restage P into A-layout scratch (wave-local; DScnt orders it)
      _Float16* pr = &Ps[wave][(hi * 8 + r) * VSTR + nl];
#pragma unroll
      for (int t = 0; t < 4; ++t) pr[t * 16] = (_Float16)e[t];
    }

    // ---- O += P V : A frags from Ps (kk chunks), B frags from Vt ----
    v16h pa0 = a_frag_lds(&Ps[wave][0], VSTR);       // kk 0..31
    v16h pa1 = a_frag_lds(&Ps[wave][0] + 32, VSTR);  // kk 32..63
#pragma unroll
    for (int t = 0; t < 4; ++t) {
      oacc[t] = wmma16(pa0, b_frag_lds(Vt,      t * 16, VSTR), oacc[t]);
      oacc[t] = wmma16(pa1, b_frag_lds(Vt + 32, t * 16, VSTR), oacc[t]);
    }
  }

  // ---- normalize and store f16 ----
  _Float16* Ob = O + (size_t)b * NQ_ * INNER_ + h * DHEAD_;
#pragma unroll
  for (int r = 0; r < 8; ++r) {
    const float inv = 1.0f / l_run[r];
    const int row = q0 + wave * 16 + hi * 8 + r;
#pragma unroll
    for (int t = 0; t < 4; ++t)
      Ob[(size_t)row * INNER_ + t * 16 + nl] = (_Float16)(oacc[t][r] * inv);
  }
}

// ---------------------------------------------------------------------------
// Host-side launch
// ---------------------------------------------------------------------------
extern "C" void kernel_launch(void* const* d_in, const int* in_sizes, int n_in,
                              void* d_out, int out_size, void* d_ws, size_t ws_size,
                              hipStream_t stream) {
  const float* x   = (const float*)d_in[0];  // [B, NQ, QDIM]
  const float* ctx = (const float*)d_in[1];  // [B, NK, CDIM]
  const float* Wq  = (const float*)d_in[2];  // [QDIM, INNER]
  const float* Wk  = (const float*)d_in[3];  // [CDIM, INNER]
  const float* Wv  = (const float*)d_in[4];  // [CDIM, INNER]
  const float* Wo  = (const float*)d_in[5];  // [INNER, QDIM]
  const float* bo  = (const float*)d_in[6];  // [QDIM]
  float* out = (float*)d_out;                // [B, NQ, QDIM] f32

  // f16 intermediates in workspace
  _Float16* qbuf = (_Float16*)d_ws;                       // [B*NQ, INNER]
  _Float16* kbuf = qbuf + (size_t)B_SZ * NQ_ * INNER_;    // [B*NK, INNER]
  _Float16* vbuf = kbuf + (size_t)B_SZ * NK_ * INNER_;    // [B*NK, INNER]
  _Float16* obuf = vbuf + (size_t)B_SZ * NK_ * INNER_;    // [B*NQ, INNER]

  const int MQ = B_SZ * NQ_;  // 16384
  const int MK = B_SZ * NK_;  // 4096

  // Projections: q = x Wq, k = ctx Wk, v = ctx Wv  (f32 in, f16 out)
  gemm_wmma_kernel<false, false, false>
      <<<dim3(MQ / 128, INNER_ / 64), 256, 0, stream>>>(x, Wq, nullptr, qbuf, MQ, QDIM_, INNER_);
  gemm_wmma_kernel<false, false, false>
      <<<dim3(MK / 128, INNER_ / 64), 256, 0, stream>>>(ctx, Wk, nullptr, kbuf, MK, CDIM_, INNER_);
  gemm_wmma_kernel<false, false, false>
      <<<dim3(MK / 128, INNER_ / 64), 256, 0, stream>>>(ctx, Wv, nullptr, vbuf, MK, CDIM_, INNER_);

  // Fused flash attention
  attn_kernel<<<dim3(NQ_ / 64, B_SZ * HEADS_), 128, 0, stream>>>(qbuf, kbuf, vbuf, obuf);

  // Output projection with bias: out = obuf Wo + bo  (f16 A, f32 out)
  gemm_wmma_kernel<true, true, true>
      <<<dim3(MQ / 128, QDIM_ / 64), 256, 0, stream>>>(obuf, Wo, bo, out, MQ, INNER_, QDIM_);
}
